// MGKVAE_49581102465149
// MI455X (gfx1250) — matrix operand order repaired and checked by reading, not compiled, not run on hardware
//
#include <hip/hip_runtime.h>
#include <hip/hip_bf16.h>

typedef _Float16 v16h __attribute__((ext_vector_type(16)));
typedef float    v8f  __attribute__((ext_vector_type(8)));

#define DATA_L   256
#define LATENT_L 64
#define NCOMP    8
#define NROWS    16384      // N = 2048 * 8
#define XROWS    2048
#define COLS_E   1536       // 3*K*L
#define COLS_D   6144       // 3*K*D
#define LDSE_STRIDE 1544    // 1536 + 8 pad
#define LDSD_STRIDE 65      // 64 + 1 pad
#define LOG2PI   1.8378770664093453f

// ---------------------------------------------------------------------------
// Pack a row-major f32 weight matrix (Krows x Ncols) into a WMMA
// 16-bit B operand in contiguous k order, split into f16 hi/lo for 3-product f32
// recovery. Layout: [(ct*nChunks + chunk)*512 + lane*16 + h], where
// K = chunk*32 + (lane>>4)*16 + h, col = ct*16 + (lane&15).
// ---------------------------------------------------------------------------
__global__ void pack_w_kernel(const float* __restrict__ W,
                              _Float16* __restrict__ hi,
                              _Float16* __restrict__ lo,
                              int Krows, int Ncols) {
  int nChunks = Krows >> 5;
  int total = (Ncols >> 4) * nChunks * 512;
  int idx = blockIdx.x * blockDim.x + threadIdx.x;
  if (idx >= total) return;
  int h    = idx & 15;
  int lane = (idx >> 4) & 31;
  int rem  = idx >> 9;
  int chunk = rem % nChunks;
  int ct    = rem / nChunks;
  int k   = (chunk << 5) + ((lane >> 4) << 4) + h;
  int col = (ct << 4) + (lane & 15);
  float f = W[(size_t)k * Ncols + col];
  _Float16 fh = (_Float16)f;
  hi[idx] = fh;
  lo[idx] = (_Float16)(f - (float)fh);
}

// ---------------------------------------------------------------------------
// Encoder: per block, 16 unique x-rows. Phase 1: pred_e = x@W_e + b_e via
// hi/lo f16 WMMA into LDS (16 x 1536). Phase 2: for the 8 replicas, sample
// mixture component, build h, accumulate P = L_q - L_e, and emit h directly
// in the packed A-matrix (hi/lo f16) layout for the decoder.
// ---------------------------------------------------------------------------
__global__ void __launch_bounds__(256)
encoder_kernel(const float* __restrict__ x,
               const _Float16* __restrict__ BeHi, const _Float16* __restrict__ BeLo,
               const float* __restrict__ b_e,
               const float* __restrict__ u_e, const float* __restrict__ r,
               _Float16* __restrict__ hHi, _Float16* __restrict__ hLo,
               float* __restrict__ P) {
  extern __shared__ float lds[];           // 16 * LDSE_STRIDE floats
  const int tid  = threadIdx.x;
  const int wave = tid >> 5;
  const int lane = tid & 31;
  const int arow = lane & 15;              // matrix row this lane feeds (A side)
  const int hs   = lane >> 4;              // half-select within K-chunk
  const float* xp = x + (size_t)(blockIdx.x * 16 + arow) * DATA_L;

  v8f acc[12] = {};
  for (int chunk = 0; chunk < 8; ++chunk) {
    // Load this lane's A fragment (two 8-float runs) and split into f16 hi/lo.
    int base = chunk * 32 + hs * 16;
    float fa[16];
#pragma unroll
    for (int i = 0; i < 8; ++i) fa[i] = xp[base + i];
#pragma unroll
    for (int i = 0; i < 8; ++i) fa[8 + i] = xp[base + 8 + i];
    v16h ahi, alo;
#pragma unroll
    for (int i = 0; i < 16; ++i) {
      _Float16 hv = (_Float16)fa[i];
      ahi[i] = hv;
      alo[i] = (_Float16)(fa[i] - (float)hv);
    }
#pragma unroll
    for (int t = 0; t < 12; ++t) {
      int ct = wave * 12 + t;
      size_t boff = ((size_t)(ct * 8 + chunk)) * 512 + lane * 16;
      v16h bhi = *(const v16h*)(BeHi + boff);
      v16h blo = *(const v16h*)(BeLo + boff);
      acc[t] = __builtin_amdgcn_wmma_f32_16x16x32_f16(false, alo, false, bhi, (short)0, acc[t], false, false);
      acc[t] = __builtin_amdgcn_wmma_f32_16x16x32_f16(false, ahi, false, blo, (short)0, acc[t], false, false);
      acc[t] = __builtin_amdgcn_wmma_f32_16x16x32_f16(false, ahi, false, bhi, (short)0, acc[t], false, false);
    }
  }
  // Bias + scatter C tiles (16x16 f32, lane = col, VGPR r = row r / r+8) to LDS.
#pragma unroll
  for (int t = 0; t < 12; ++t) {
    int col  = (wave * 12 + t) * 16 + (lane & 15);
    float bias = b_e[col];
#pragma unroll
    for (int rr = 0; rr < 8; ++rr) {
      int mrow = rr + (hs << 3);
      lds[mrow * LDSE_STRIDE + col] = acc[t][rr] + bias;
    }
  }
  __syncthreads();

  // Phase 2: sampling + L_e/L_q for the 8 replicas of these 16 rows.
  if (tid < 128) {
    int rep  = tid >> 4;
    int srow = tid & 15;
    int n    = rep * XROWS + blockIdx.x * 16 + srow;
    int rt   = rep * 128 + blockIdx.x;     // decoder A row-tile index
    const float* up = u_e + (size_t)n * LATENT_L;
    const float* rp = r   + (size_t)n * LATENT_L;
    const float* pe = lds + srow * LDSE_STRIDE;
    float sumH2 = 0.f, sumLe = 0.f;
    for (int l = 0; l < LATENT_L; ++l) {
      float m[8], lp[8], la[8], lan[8], tv[8];
#pragma unroll
      for (int k = 0; k < 8; ++k) {
        m[k]  = pe[k * 64 + l];
        lp[k] = pe[512 + k * 64 + l];
        la[k] = pe[1024 + k * 64 + l];
      }
      float mx = la[0];
#pragma unroll
      for (int k = 1; k < 8; ++k) mx = fmaxf(mx, la[k]);
      float s = 0.f;
#pragma unroll
      for (int k = 0; k < 8; ++k) s += __expf(la[k] - mx);
      float lse = mx + __logf(s);
      // discrete sample: idx = sum(r > cumsum(softmax)), clipped
      float rv = rp[l];
      float c = 0.f;
      int idx = 0;
#pragma unroll
      for (int k = 0; k < 8; ++k) {
        lan[k] = la[k] - lse;
        c += __expf(lan[k]);
        idx += (rv > c) ? 1 : 0;
      }
      idx = idx > 7 ? 7 : idx;
      float msel = m[0], lpsel = lp[0];
#pragma unroll
      for (int k = 1; k < 8; ++k) {
        bool take = (idx == k);
        msel  = take ? m[k]  : msel;
        lpsel = take ? lp[k] : lpsel;
      }
      float hv = msel + __expf(-0.5f * lpsel) * up[l];
      sumH2 += hv * hv;
      // L_e contribution: logsumexp_k(lan + 0.5*lp - 0.5*(exp(0.5*lp)*(h-m))^2)
      float mx2 = -1e30f;
#pragma unroll
      for (int k = 0; k < 8; ++k) {
        float u = __expf(0.5f * lp[k]) * (hv - m[k]);
        tv[k] = lan[k] + 0.5f * lp[k] - 0.5f * u * u;
        mx2 = fmaxf(mx2, tv[k]);
      }
      float s2 = 0.f;
#pragma unroll
      for (int k = 0; k < 8; ++k) s2 += __expf(tv[k] - mx2);
      sumLe += mx2 + __logf(s2);
      // Emit h in packed A-matrix layout (hi/lo f16): chunk = l>>5
      _Float16 hh = (_Float16)hv;
      _Float16 hl = (_Float16)(hv - (float)hh);
      int ch  = l >> 5;
      int lq  = l & 31;
      int alane = srow + ((lq >> 4) ? 16 : 0);
      int ahalf = lq & 15;
      size_t aoff = ((size_t)(rt * 2 + ch) * 32 + alane) * 16 + ahalf;
      hHi[aoff] = hh;
      hLo[aoff] = hl;
    }
    P[n] = -0.5f * sumH2 - sumLe;          // L_q - L_e (log2pi terms cancel)
  }
}

// ---------------------------------------------------------------------------
// Decoder: block = (row-tile rt of 16 rows) x (d-group g of 64 data dims).
// WMMA h@W_d for all 3 parts x 8 comps x 64 d (96 col-tiles) into LDS, then
// fused log-softmax + logsumexp_k + sum_d reduction. No pred_d to HBM.
// ---------------------------------------------------------------------------
__global__ void __launch_bounds__(256)
decoder_kernel(const float* __restrict__ x,
               const _Float16* __restrict__ BdHi, const _Float16* __restrict__ BdLo,
               const float* __restrict__ b_d,
               const _Float16* __restrict__ hHi, const _Float16* __restrict__ hLo,
               const float* __restrict__ P,
               float* __restrict__ partials) {
  extern __shared__ float lds[];           // 24*16*LDSD_STRIDE + 256 floats
  float* red = lds + 24 * 16 * LDSD_STRIDE;
  const int tid  = threadIdx.x;
  const int wave = tid >> 5;
  const int lane = tid & 31;
  const int rt = blockIdx.x >> 2;          // row tile (16 rows of h)
  const int g  = blockIdx.x & 3;           // group of 64 data dims

  // A fragments (h, packed by encoder): K=64 -> 2 chunks, hi/lo.
  v16h ahi[2], alo[2];
#pragma unroll
  for (int c = 0; c < 2; ++c) {
    size_t off = ((size_t)(rt * 2 + c) * 32 + lane) * 16;
    ahi[c] = *(const v16h*)(hHi + off);
    alo[c] = *(const v16h*)(hLo + off);
  }

#pragma unroll
  for (int t = 0; t < 12; ++t) {
    int tt  = wave * 12 + t;               // 96 col-tiles / block
    int rix = tt >> 2, sub = tt & 3;
    int p = rix >> 3, k = rix & 7;         // part (m/logp/loga), component
    int ct = p * 128 + k * 16 + g * 4 + sub;   // global 16-col tile of pred_d
    v8f cacc = {};
#pragma unroll
    for (int c2 = 0; c2 < 2; ++c2) {
      size_t boff = ((size_t)(ct * 2 + c2)) * 512 + lane * 16;
      v16h bhi = *(const v16h*)(BdHi + boff);
      v16h blo = *(const v16h*)(BdLo + boff);
      cacc = __builtin_amdgcn_wmma_f32_16x16x32_f16(false, alo[c2], false, bhi, (short)0, cacc, false, false);
      cacc = __builtin_amdgcn_wmma_f32_16x16x32_f16(false, ahi[c2], false, blo, (short)0, cacc, false, false);
      cacc = __builtin_amdgcn_wmma_f32_16x16x32_f16(false, ahi[c2], false, bhi, (short)0, cacc, false, false);
    }
    int colLocal = lane & 15;
    float bias = b_d[ct * 16 + colLocal];
    int dlocal = sub * 16 + colLocal;
#pragma unroll
    for (int rr = 0; rr < 8; ++rr) {
      int mrow = rr + ((lane >> 4) << 3);
      lds[((p * 8 + k) * 16 + mrow) * LDSD_STRIDE + dlocal] = cacc[rr] + bias;
    }
  }
  __syncthreads();

  // Fused mixture-likelihood reduction over the LDS slab.
  float partial = 0.f;
  for (int task = tid; task < 1024; task += 256) {
    int mrow = task >> 6, dl = task & 63;
    int n = rt * 16 + mrow;
    float xv = x[(size_t)(n & (XROWS - 1)) * DATA_L + g * 64 + dl];
    float m[8], lp[8], la[8], tv[8];
#pragma unroll
    for (int k = 0; k < 8; ++k) {
      m[k]  = lds[((0 * 8 + k) * 16 + mrow) * LDSD_STRIDE + dl];
      lp[k] = lds[((1 * 8 + k) * 16 + mrow) * LDSD_STRIDE + dl];
      la[k] = lds[((2 * 8 + k) * 16 + mrow) * LDSD_STRIDE + dl];
    }
    float mx = la[0];
#pragma unroll
    for (int k = 1; k < 8; ++k) mx = fmaxf(mx, la[k]);
    float s = 0.f;
#pragma unroll
    for (int k = 0; k < 8; ++k) s += __expf(la[k] - mx);
    float lseA = mx + __logf(s);
    float mx2 = -1e30f;
#pragma unroll
    for (int k = 0; k < 8; ++k) {
      float u = __expf(0.5f * lp[k]) * (xv - m[k]);
      tv[k] = (la[k] - lseA) + 0.5f * lp[k] - 0.5f * u * u;
      mx2 = fmaxf(mx2, tv[k]);
    }
    float s2 = 0.f;
#pragma unroll
    for (int k = 0; k < 8; ++k) s2 += __expf(tv[k] - mx2);
    partial += mx2 + __logf(s2);
  }
  red[tid] = partial;
  __syncthreads();
  for (int st = 128; st > 0; st >>= 1) {
    if (tid < st) red[tid] += red[tid + st];
    __syncthreads();
  }
  if (tid == 0) {
    float tot = red[0];
    if (g == 0) {
      for (int rr = 0; rr < 16; ++rr) tot += P[rt * 16 + rr];
    }
    partials[blockIdx.x] = tot;            // deterministic per-block partial
  }
}

__global__ void final_reduce_kernel(const float* __restrict__ partials,
                                    float* __restrict__ out) {
  __shared__ float red[256];
  int tid = threadIdx.x;
  float s = 0.f;
  for (int i = 0; i < 16; ++i) s += partials[tid + 256 * i];
  red[tid] = s;
  __syncthreads();
  for (int st = 128; st > 0; st >>= 1) {
    if (tid < st) red[tid] += red[tid + st];
    __syncthreads();
  }
  if (tid == 0)
    out[0] = 0.5f * (float)DATA_L * LOG2PI - red[0] * (1.0f / (float)NROWS);
}

// ---------------------------------------------------------------------------
extern "C" void kernel_launch(void* const* d_in, const int* in_sizes, int n_in,
                              void* d_out, int out_size, void* d_ws, size_t ws_size,
                              hipStream_t stream) {
  const float* x   = (const float*)d_in[0];
  const float* W_e = (const float*)d_in[1];
  const float* b_e = (const float*)d_in[2];
  const float* W_d = (const float*)d_in[3];
  const float* b_d = (const float*)d_in[4];
  const float* u_e = (const float*)d_in[5];
  const float* r   = (const float*)d_in[6];

  char* ws = (char*)d_ws;
  size_t off = 0;
  auto suballoc = [&](size_t bytes) -> void* {
    void* p = ws + off;
    off += (bytes + 255) & ~(size_t)255;
    return p;
  };
  _Float16* BeHi = (_Float16*)suballoc((size_t)DATA_L * COLS_E * 2);
  _Float16* BeLo = (_Float16*)suballoc((size_t)DATA_L * COLS_E * 2);
  _Float16* BdHi = (_Float16*)suballoc((size_t)LATENT_L * COLS_D * 2);
  _Float16* BdLo = (_Float16*)suballoc((size_t)LATENT_L * COLS_D * 2);
  _Float16* hHi  = (_Float16*)suballoc((size_t)NROWS * LATENT_L * 2);
  _Float16* hLo  = (_Float16*)suballoc((size_t)NROWS * LATENT_L * 2);
  float*    P    = (float*)suballoc((size_t)NROWS * 4);
  float*    part = (float*)suballoc((size_t)4096 * 4);

  int packElems = (COLS_E / 16) * (DATA_L / 32) * 512;   // 393216 (same for W_d)
  pack_w_kernel<<<(packElems + 255) / 256, 256, 0, stream>>>(W_e, BeHi, BeLo, DATA_L, COLS_E);
  pack_w_kernel<<<(packElems + 255) / 256, 256, 0, stream>>>(W_d, BdHi, BdLo, LATENT_L, COLS_D);

  size_t ldsE = (size_t)16 * LDSE_STRIDE * sizeof(float);            // ~98.8 KB
  encoder_kernel<<<128, 256, ldsE, stream>>>(x, BeHi, BeLo, b_e, u_e, r, hHi, hLo, P);

  size_t ldsD = ((size_t)24 * 16 * LDSD_STRIDE + 256) * sizeof(float); // ~100.8 KB
  decoder_kernel<<<4096, 256, ldsD, stream>>>(x, BdHi, BdLo, b_d, hHi, hLo, P, part);

  final_reduce_kernel<<<1, 256, 0, stream>>>(part, (float*)d_out);
}
